// GroupedQueryAttention_58222576664630
// MI455X (gfx1250) — compile-verified, matrix operand order checked by reading, not run on hardware
//
#include <hip/hip_runtime.h>
#include <math.h>

// ---------------- types / constants ----------------
typedef __bf16         bf16x16 __attribute__((ext_vector_type(16)));
typedef float          f32x8   __attribute__((ext_vector_type(8)));
typedef unsigned short u16x8   __attribute__((ext_vector_type(8)));
typedef unsigned int   u32x4   __attribute__((ext_vector_type(4)));
typedef int            i32x4   __attribute__((ext_vector_type(4)));
typedef int            i32x8   __attribute__((ext_vector_type(8)));

#define D_MODEL   2048
#define N_HEADS_C 16
#define N_KV_C    4
#define HD        128
#define SEQ       2048
#define BATCH     2
#define TOKENS    (BATCH * SEQ)
#define KV_DIM    (N_KV_C * HD)   // 512

// ---------------- helpers ----------------
__device__ __forceinline__ unsigned short f32_to_bf16(float f) {
  unsigned int u = __float_as_uint(f);
  u += 0x7FFFu + ((u >> 16) & 1u);          // round-to-nearest-even
  return (unsigned short)(u >> 16);
}

__device__ __forceinline__ f32x8 wmma_bf16(bf16x16 a, bf16x16 b, f32x8 c) {
  return __builtin_amdgcn_wmma_f32_16x16x32_bf16(
      /*neg_a=*/false, a, /*neg_b=*/false, b,
      /*c_mod=*/(short)0, c, /*reuse_a=*/false, /*reuse_b=*/false);
}

// CDNA5 async copy: per-lane 16B global -> LDS, tracked by ASYNCcnt.
// Low 32 bits of a generic LDS pointer are the LDS byte address (ISA 10.2).
__device__ __forceinline__ void async_load_b128(void* lds, const void* gptr) {
  unsigned l = (unsigned)(unsigned long long)lds;
  asm volatile("global_load_async_to_lds_b128 %0, %1, off"
               :: "v"(l), "v"(gptr) : "memory");
}
__device__ __forceinline__ void wait_async0() {
  asm volatile("s_wait_asynccnt 0x0" ::: "memory");
}

// Tensor Data Mover: DMA one 128x128 bf16 tile (row stride `stride_elems`)
// into LDS with pitch 136 elems (pad 4 dwords after every 64 dwords = one row).
// D# layout per CDNA5 ISA 8.3/8.4 (group0 + group1, 2D, groups 2/3 zero).
// This toolchain exposes the 6-arg builtin (4 descriptor groups + spare + cpol).
__device__ __forceinline__ void tdm_load_tile_128x128(void* lds, const void* gptr,
                                                      int stride_elems) {
  unsigned long long ga = (unsigned long long)gptr;
  unsigned la = (unsigned)(unsigned long long)lds;
  u32x4 g0;
  g0.x = 1u;                                                  // count=1, user desc
  g0.y = la;                                                  // lds_addr (bytes)
  g0.z = (unsigned)(ga & 0xffffffffu);                        // global_addr[31:0]
  g0.w = (unsigned)((ga >> 32) & 0x01ffffffu) | (2u << 30);   // ga[56:32] | type=2
  i32x8 g1;
  g1[0] = (int)((1u << 16)        // data_size = 2 bytes
              | (1u << 20)        // pad_enable
              | (5u << 22)        // pad_interval: 64 dwords (one 128-elem row)
              | (3u << 25));      // pad_amount:   4 dwords (8 elems) -> pitch 136
  g1[1] = (int)(128u << 16);      // tensor_dim0 lo16 = 128 (atomic_barrier_addr = 0)
  g1[2] = (int)(0x4000u << 16);   // tensor_dim0 hi = 0 ; tensor_dim1 lo16 = 16384
  g1[3] = (int)(128u << 16);      // tensor_dim1 hi = 0 ; tile_dim0 = 128
  g1[4] = 128;                    // tile_dim1 = 128 ; tile_dim2 = 0 (2D)
  g1[5] = stride_elems;           // tensor_dim0_stride[31:0] (data_size units)
  g1[6] = 0;
  g1[7] = 0;
  i32x4 z4 = {0, 0, 0, 0};
  i32x8 z8 = {0, 0, 0, 0, 0, 0, 0, 0};
  __builtin_amdgcn_tensor_load_to_lds(g0, g1, z4, z4, z8, 0);
}

// 16-bit A/B WMMA fragment from LDS (CDNA5 operand layout):
// lanes 0-15: row row16+lane, K = kbase+0..7 / +16..23 ; lanes 16-31: +8.
__device__ __forceinline__ bf16x16 lds_frag(const unsigned short* t, int stride,
                                            int row16, int kbase, int lane) {
  const int r = row16 + (lane & 15);
  const int k = kbase + ((lane >> 4) << 3);
  union { bf16x16 v; u16x8 h[2]; } u;
  const unsigned short* p = t + r * stride + k;
  u.h[0] = *(const u16x8*)p;         // ds_load_b128
  u.h[1] = *(const u16x8*)(p + 16);  // ds_load_b128
  return u.v;
}

// ---------------- elementwise kernels ----------------
__global__ void k_convert_bf16(const float* __restrict__ in,
                               unsigned short* __restrict__ out, int n) {
  int i = blockIdx.x * blockDim.x + threadIdx.x;
  if (i < n) out[i] = f32_to_bf16(in[i]);
}

// W [K,N] row-major f32 -> Wt [N,K] row-major bf16 (K-contiguous for WMMA B)
__global__ void k_transpose_bf16(const float* __restrict__ W,
                                 unsigned short* __restrict__ Wt, int K, int N) {
  int i = blockIdx.x * blockDim.x + threadIdx.x;
  if (i >= K * N) return;
  int k = i / N, n = i - k * N;
  Wt[(size_t)n * K + k] = f32_to_bf16(W[i]);
}

// Reference quirk: RoPE angle is indexed by the HEAD axis (per-head rotation,
// position-independent).  in: [TOKENS, heads, HD] f32 -> bf16 rotated.
__global__ void k_rope_bf16(const float* __restrict__ in,
                            unsigned short* __restrict__ out, int heads) {
  int idx = blockIdx.x * blockDim.x + threadIdx.x;        // pair index
  int total = TOKENS * heads * (HD / 2);
  if (idx >= total) return;
  int i = idx & (HD / 2 - 1);                              // 0..63
  int h = (idx >> 6) % heads;
  float ang = (float)h * __powf(10000.0f, -(float)i * (1.0f / 64.0f));
  float s, c;
  __sincosf(ang, &s, &c);
  const float* p = in + (size_t)idx * 2;
  float x0 = p[0], x1 = p[1];
  out[(size_t)idx * 2]     = f32_to_bf16(x0 * c - x1 * s);
  out[(size_t)idx * 2 + 1] = f32_to_bf16(x0 * s + x1 * c);
}

// ---------------- double-buffered async WMMA GEMM ----------------
// C[M,N] f32 = A[M,K] bf16 * Bt[N,K] bf16 ; block 128x256, wave tile 64x64.
#define BM 128
#define BN 256
#define BK 32
#define LDK (BK + 8)   // pad: keeps 16B alignment, avoids bank conflicts

__global__ __launch_bounds__(256)
void gemm_bf16_nt(const unsigned short* __restrict__ A,
                  const unsigned short* __restrict__ Bt,
                  float* __restrict__ C, int M, int N, int K) {
  (void)M;
  __shared__ unsigned short As[2][BM * LDK];   // 2 x 10 KB
  __shared__ unsigned short Bs[2][BN * LDK];   // 2 x 20 KB
  const int tid  = threadIdx.x;
  const int lane = tid & 31;
  const int wid  = tid >> 5;
  const int wm   = wid >> 2;            // 2 waves along M
  const int wn   = wid & 3;             // 4 waves along N
  const int bm   = blockIdx.y * BM;
  const int bn   = blockIdx.x * BN;

  f32x8 acc[4][4] = {};                 // wave tile 64x64 = 4x4 16x16 tiles

  auto stage = [&](int buf, int k0) {
    #pragma unroll
    for (int i = 0; i < (BM * BK) / 8 / 256; ++i) {       // 2 async per lane
      const int idx = tid + i * 256;
      const int r = idx >> 2, c = (idx & 3) << 3;
      async_load_b128(&As[buf][r * LDK + c], &A[(size_t)(bm + r) * K + k0 + c]);
    }
    #pragma unroll
    for (int i = 0; i < (BN * BK) / 8 / 256; ++i) {       // 4 async per lane
      const int idx = tid + i * 256;
      const int r = idx >> 2, c = (idx & 3) << 3;
      async_load_b128(&Bs[buf][r * LDK + c], &Bt[(size_t)(bn + r) * K + k0 + c]);
    }
  };

  stage(0, 0);
  int buf = 0;
  for (int k0 = 0; k0 < K; k0 += BK) {
    wait_async0();                       // this wave's portion of tile[buf] landed
    __syncthreads();                     // all waves done staging + prev compute
    if (k0 + BK < K) stage(buf ^ 1, k0 + BK);   // overlap next tile with compute

    bf16x16 af[4];
    #pragma unroll
    for (int mt = 0; mt < 4; ++mt)
      af[mt] = lds_frag(&As[buf][0], LDK, wm * 64 + mt * 16, 0, lane);
    #pragma unroll
    for (int nt = 0; nt < 4; ++nt) {
      bf16x16 bfr = lds_frag(&Bs[buf][0], LDK, wn * 64 + nt * 16, 0, lane);
      #pragma unroll
      for (int mt = 0; mt < 4; ++mt)
        acc[mt][nt] = wmma_bf16(af[mt], bfr, acc[mt][nt]);
    }
    buf ^= 1;
  }

  const int hi8 = (lane >> 4) << 3;     // C layout: vgpr r = row r / r+8 per lane half
  #pragma unroll
  for (int mt = 0; mt < 4; ++mt)
    #pragma unroll
    for (int nt = 0; nt < 4; ++nt)
      #pragma unroll
      for (int r = 0; r < 8; ++r) {
        const int row = bm + wm * 64 + mt * 16 + r + hi8;
        const int col = bn + wn * 64 + nt * 16 + (lane & 15);
        C[(size_t)row * N + col] = acc[mt][nt][r];
      }
}

// ---------------- flash attention (causal, GQA) ----------------
#define BQ  128
#define BKV 128
#define LDT (HD + 8)      // 136 — matches the TDM pad descriptor
#define LKV (BKV + 8)     // 136

__global__ __launch_bounds__(256)
void attn_fwd(const unsigned short* __restrict__ Qb,   // [TOKENS,16,HD] bf16
              const unsigned short* __restrict__ Kb,   // [TOKENS, 4,HD] bf16
              const unsigned short* __restrict__ Vb,   // [TOKENS, 4,HD] bf16
              unsigned short* __restrict__ Ob) {       // [TOKENS, 2048] bf16
  extern __shared__ unsigned short smem[];
  unsigned short* Qs = smem;               // [BQ ][LDT]
  unsigned short* Ks = Qs + BQ * LDT;      // [BKV][LDT]  key rows, hd contiguous
  unsigned short* Vt = Ks + BKV * LDT;     // [HD ][LKV]  V transposed
  unsigned short* Ps = Vt + HD * LKV;      // [BQ ][LKV]  probabilities

  const int tid = threadIdx.x, lane = tid & 31, wid = tid >> 5;
  const int qb  = blockIdx.x;              // 128-row query block
  const int bh  = blockIdx.y;              // b*16 + h
  const int b   = bh >> 4;
  const int h   = bh & 15;
  const int g   = h >> 2;                  // kv group = h / N_REP
  const float scale = 0.088388347648318447f;   // 1/sqrt(128)
  const int hi8 = (lane >> 4) << 3;

  // stage Q block via Tensor Data Mover (wave 0 issues, everyone syncs)
  if (wid == 0) {
    tdm_load_tile_128x128(
        Qs, &Qb[(((size_t)(b * SEQ + qb * BQ)) * N_HEADS_C + h) * HD],
        N_HEADS_C * HD);
    __builtin_amdgcn_s_wait_tensorcnt(0);
  }
  __syncthreads();

  bf16x16 qf[4];                           // wave's 16 query rows, full hd
  #pragma unroll
  for (int ks = 0; ks < 4; ++ks) qf[ks] = lds_frag(Qs, LDT, wid * 16, ks * 32, lane);

  f32x8 oacc[8] = {};
  float m_r[8], l_r[8];
  #pragma unroll
  for (int r = 0; r < 8; ++r) { m_r[r] = -1e30f; l_r[r] = 0.0f; }

  for (int j = 0; j <= qb; ++j) {          // causal: chunks up to the diagonal
    // K chunk via TDM; V chunk transposed manually (TDM cannot transpose)
    if (wid == 0)
      tdm_load_tile_128x128(
          Ks, &Kb[(((size_t)(b * SEQ + j * BKV)) * N_KV_C + g) * HD],
          N_KV_C * HD);
    for (int i = tid; i < (BKV * HD) / 8; i += 256) {
      const int r = i >> 4, c = (i & 15) << 3;
      const int t = j * BKV + r;
      u16x8 v = *(const u16x8*)&Vb[(((size_t)(b * SEQ + t)) * N_KV_C + g) * HD + c];
      #pragma unroll
      for (int e = 0; e < 8; ++e) Vt[(c + e) * LKV + r] = v[e];
    }
    if (wid == 0) __builtin_amdgcn_s_wait_tensorcnt(0);
    __syncthreads();

    // S = Q * K^T  (wave: 16 q-rows x 128 keys = 8 tiles, 4 k-steps over hd)
    f32x8 sacc[8];
    #pragma unroll
    for (int nt = 0; nt < 8; ++nt) {
      f32x8 s = {};
      #pragma unroll
      for (int ks = 0; ks < 4; ++ks) {
        bf16x16 kf = lds_frag(Ks, LDT, nt * 16, ks * 32, lane);
        s = wmma_bf16(qf[ks], kf, s);
      }
      sacc[nt] = s;
    }

    // scale + causal mask + online softmax
    const bool diag = (j == qb);
    const int  keyc = lane & 15;
    #pragma unroll
    for (int r = 0; r < 8; ++r) {
      const int qloc = wid * 16 + r + hi8;
      float mx = -1e30f;
      #pragma unroll
      for (int nt = 0; nt < 8; ++nt) {
        float sv = sacc[nt][r] * scale;
        if (diag && (nt * 16 + keyc) > qloc) sv = -1e30f;
        sacc[nt][r] = sv;
        mx = fmaxf(mx, sv);
      }
      #pragma unroll
      for (int d = 8; d >= 1; d >>= 1) mx = fmaxf(mx, __shfl_xor(mx, d, 16));
      const float mnew  = fmaxf(m_r[r], mx);
      const float alpha = __expf(m_r[r] - mnew);
      float lsum = 0.0f;
      #pragma unroll
      for (int nt = 0; nt < 8; ++nt) {
        float p = __expf(sacc[nt][r] - mnew);
        sacc[nt][r] = p;
        lsum += p;
      }
      #pragma unroll
      for (int d = 8; d >= 1; d >>= 1) lsum += __shfl_xor(lsum, d, 16);
      l_r[r] = l_r[r] * alpha + lsum;
      m_r[r] = mnew;
      #pragma unroll
      for (int ot = 0; ot < 8; ++ot) oacc[ot][r] *= alpha;
      #pragma unroll
      for (int nt = 0; nt < 8; ++nt)
        Ps[(size_t)qloc * LKV + nt * 16 + keyc] = f32_to_bf16(sacc[nt][r]);
    }
    __syncthreads();

    // O += P * V   (A = this wave's P rows, B^T = Vt)
    bf16x16 pf[4];
    #pragma unroll
    for (int ks = 0; ks < 4; ++ks) pf[ks] = lds_frag(Ps, LKV, wid * 16, ks * 32, lane);
    #pragma unroll
    for (int ot = 0; ot < 8; ++ot) {
      #pragma unroll
      for (int ks = 0; ks < 4; ++ks) {
        bf16x16 vf = lds_frag(Vt, LKV, ot * 16, ks * 32, lane);
        oacc[ot] = wmma_bf16(pf[ks], vf, oacc[ot]);
      }
    }
    __syncthreads();
  }

  // normalize and emit bf16 for the Wo GEMM
  #pragma unroll
  for (int r = 0; r < 8; ++r) {
    const float inv_l = 1.0f / l_r[r];
    const int row = qb * BQ + wid * 16 + r + hi8;
    #pragma unroll
    for (int ot = 0; ot < 8; ++ot) {
      const int col = h * HD + ot * 16 + (lane & 15);
      Ob[(size_t)(b * SEQ + row) * D_MODEL + col] = f32_to_bf16(oacc[ot][r] * inv_l);
    }
  }
}

// ---------------- launcher ----------------
extern "C" void kernel_launch(void* const* d_in, const int* in_sizes, int n_in,
                              void* d_out, int out_size, void* d_ws, size_t ws_size,
                              hipStream_t stream) {
  (void)in_sizes; (void)n_in; (void)out_size; (void)ws_size;
  const float* x  = (const float*)d_in[0];
  const float* Wq = (const float*)d_in[1];
  const float* Wk = (const float*)d_in[2];
  const float* Wv = (const float*)d_in[3];
  const float* Wo = (const float*)d_in[4];
  // d_in[5] = attention_mask (causal) -- regenerated analytically in-kernel
  float* out = (float*)d_out;

  char* ws = (char*)d_ws;
  size_t off = 0;
  auto take = [&](size_t bytes) -> char* {
    char* p = ws + off;
    off += (bytes + 255) & ~(size_t)255;
    return p;
  };
  unsigned short* xb    = (unsigned short*)take((size_t)TOKENS * D_MODEL * 2);
  unsigned short* WqT   = (unsigned short*)take((size_t)D_MODEL * D_MODEL * 2);
  unsigned short* WkT   = (unsigned short*)take((size_t)KV_DIM * D_MODEL * 2);
  unsigned short* WvT   = (unsigned short*)take((size_t)KV_DIM * D_MODEL * 2);
  unsigned short* WoT   = (unsigned short*)take((size_t)D_MODEL * D_MODEL * 2);
  unsigned short* qb16  = (unsigned short*)take((size_t)TOKENS * D_MODEL * 2);
  unsigned short* kb16  = (unsigned short*)take((size_t)TOKENS * KV_DIM * 2);
  unsigned short* vb16  = (unsigned short*)take((size_t)TOKENS * KV_DIM * 2);
  unsigned short* attnb = (unsigned short*)take((size_t)TOKENS * D_MODEL * 2);
  float*          f32t  = (float*)take((size_t)TOKENS * D_MODEL * 4);   // reused scratch

  const int T = 256;
  k_convert_bf16<<<(TOKENS * D_MODEL + T - 1) / T, T, 0, stream>>>(x, xb, TOKENS * D_MODEL);
  k_transpose_bf16<<<(D_MODEL * D_MODEL + T - 1) / T, T, 0, stream>>>(Wq, WqT, D_MODEL, D_MODEL);
  k_transpose_bf16<<<(D_MODEL * KV_DIM + T - 1) / T, T, 0, stream>>>(Wk, WkT, D_MODEL, KV_DIM);
  k_transpose_bf16<<<(D_MODEL * KV_DIM + T - 1) / T, T, 0, stream>>>(Wv, WvT, D_MODEL, KV_DIM);
  k_transpose_bf16<<<(D_MODEL * D_MODEL + T - 1) / T, T, 0, stream>>>(Wo, WoT, D_MODEL, D_MODEL);

  // Q projection + per-head RoPE
  gemm_bf16_nt<<<dim3(D_MODEL / BN, TOKENS / BM), T, 0, stream>>>(xb, WqT, f32t, TOKENS, D_MODEL, D_MODEL);
  k_rope_bf16<<<(TOKENS * N_HEADS_C * (HD / 2) + T - 1) / T, T, 0, stream>>>(f32t, qb16, N_HEADS_C);
  // K projection + per-head RoPE
  gemm_bf16_nt<<<dim3(KV_DIM / BN, TOKENS / BM), T, 0, stream>>>(xb, WkT, f32t, TOKENS, KV_DIM, D_MODEL);
  k_rope_bf16<<<(TOKENS * N_KV_C * (HD / 2) + T - 1) / T, T, 0, stream>>>(f32t, kb16, N_KV_C);
  // V projection
  gemm_bf16_nt<<<dim3(KV_DIM / BN, TOKENS / BM), T, 0, stream>>>(xb, WvT, f32t, TOKENS, KV_DIM, D_MODEL);
  k_convert_bf16<<<(TOKENS * KV_DIM + T - 1) / T, T, 0, stream>>>(f32t, vb16, TOKENS * KV_DIM);

  // flash attention: grid = (q blocks, batch*heads); ~136KB dynamic LDS (<320KB WGP)
  const size_t smem = (size_t)(BQ * LDT + BKV * LDT + HD * LKV + BQ * LKV) * 2;
  attn_fwd<<<dim3(SEQ / BQ, BATCH * N_HEADS_C), T, smem, stream>>>(qb16, kb16, vb16, attnb);

  // output projection -> f32 result
  gemm_bf16_nt<<<dim3(D_MODEL / BN, TOKENS / BM), T, 0, stream>>>(attnb, WoT, out, TOKENS, D_MODEL, D_MODEL);
}
